// CSEM_86294482911613
// MI455X (gfx1250) — matrix operand order repaired and checked by reading, not compiled, not run on hardware
//
#include <hip/hip_runtime.h>
#include <hip/hip_bf16.h>
#include <math.h>

// ---------------------------------------------------------------------------
// MI455X (gfx1250) implementation.
// Heavy ops (conv0, conv1, q@k^T, attn@v) are tiled implicit-GEMM kernels on
// v_wmma_f32_16x16x32_f16 (f16 operands, f32 accumulate):
//  - 64x128 tiles (conv / attn-v), each wave32 computes a 32x32 block
//    (4 WMMAs per K-chunk from 8 b128 fragment loads).
//  - software pipelined: global gathers for chunk k+1 issue before the WMMAs
//    of chunk k, so VMEM latency hides under matrix ops.
//  - double-buffered LDS tiles: ONE workgroup barrier per K-chunk.
//  - all LDS staging stores are b128 (packed f16 pairs via v_cvt_pk_f16_f32).
//  - halo predication is branchless (cndmask address clamp + select-zero),
//    no per-element exec-mask branches.
// ---------------------------------------------------------------------------

typedef __attribute__((ext_vector_type(16))) _Float16 v16h;
typedef __attribute__((ext_vector_type(8)))  _Float16 v8h;
typedef __attribute__((ext_vector_type(8)))  float    v8f;

#define LSTR 40   // LDS row stride in halves (80B, 16B-aligned, conflict-friendly)

// A-operand fragment (16x32 f16): lanes 0-15 hold K=0..15 contiguous, lanes
// 16-31 hold K=16..31. => two b128 loads at +0 / +8.
__device__ __forceinline__ v16h frag_a(const _Float16* p) {
  v8h lo = *(const v8h*)(p);
  v8h hi = *(const v8h*)(p + 8);
  return __builtin_shufflevector(lo, hi, 0,1,2,3,4,5,6,7,8,9,10,11,12,13,14,15);
}
// B-operand fragment (32x16 f16): lanes 0-15 hold K=0..15 contiguous, lanes
// 16-31 hold K=16..31. => 16 contiguous halves (two b128 loads).
__device__ __forceinline__ v16h frag_b(const _Float16* p) {
  v8h lo = *(const v8h*)(p);
  v8h hi = *(const v8h*)(p + 8);
  return __builtin_shufflevector(lo, hi, 0,1,2,3,4,5,6,7,8,9,10,11,12,13,14,15);
}

__device__ __forceinline__ v8f wmma_f16(v16h a, v16h b, v8f c) {
  return __builtin_amdgcn_wmma_f32_16x16x32_f16(false, a, false, b, (short)0, c,
                                                false, false);
}

// Convert 8 f32 -> 8 contiguous f16 and store as one b128.
__device__ __forceinline__ void store_pk8(_Float16* dst, const float* src) {
  v8h v;
#pragma unroll
  for (int j = 0; j < 8; ++j) v[j] = (_Float16)src[j];
  *(v8h*)dst = v;
}

// ---------------------------------------------------------------------------
// K1: fold BN into scale/bias (absorbs conv bias).
// par layout: [0,192) s0 | [192,384) b0f | [384,960) s1 | [960,1536) b1f
// ---------------------------------------------------------------------------
__global__ void fold_bn(const float* __restrict__ b0, const float* __restrict__ g0,
                        const float* __restrict__ be0, const float* __restrict__ m0,
                        const float* __restrict__ v0,
                        const float* __restrict__ b1, const float* __restrict__ g1,
                        const float* __restrict__ be1, const float* __restrict__ m1,
                        const float* __restrict__ v1, float* __restrict__ par) {
  int i = blockIdx.x * 256 + threadIdx.x;
  if (i < 192) {
    float s = g0[i] * rsqrtf(v0[i] + 1e-5f);
    par[i] = s;
    par[192 + i] = be0[i] + (b0[i] - m0[i]) * s;
  } else if (i < 768) {
    int j = i - 192;
    float s = g1[j] * rsqrtf(v1[j] + 1e-5f);
    par[384 + j] = s;
    par[960 + j] = be1[j] + (b1[j] - m1[j]) * s;
  }
}

// ---------------------------------------------------------------------------
// K2/K4: 3x3 conv (pad 1) as implicit GEMM + fused BN + ReLU.
//   D[Cout][B*H*W] = W[Cout][Cin*9] x Im2col[Cin*9][B*H*W]
// 256 threads (8 wave32), 64(M) x 128(N) tile, K chunks of 32, pipelined,
// double-buffered LDS. Waves: wm = wave&1 (32 rows), wn = wave>>1 (32 cols).
// ---------------------------------------------------------------------------
__global__ __launch_bounds__(256) void conv3x3_wmma(
    const float* __restrict__ X, const float* __restrict__ Wt,
    const float* __restrict__ scale, const float* __restrict__ bias,
    float* __restrict__ Y, int Cin, int Cout) {
  __shared__ _Float16 As[2][64 * LSTR];
  __shared__ _Float16 Bs[2][128 * LSTR];
  const int bz   = blockIdx.z;
  const int m0   = blockIdx.y * 64;
  const int col0 = blockIdx.x * 128;
  const int Ktot = Cin * 9;
  const int tid  = threadIdx.x;
  const int lane = tid & 31;
  const int wave = tid >> 5;
  const int wm = wave & 1, wn = wave >> 1;
  const int lrow = lane & 15;
  const int asel = (lane & 16) ? 16 : 0;
  const int bsel = (lane & 16) ? 16 : 0;

  // A staging slot: 64 rows x 32 k, 8 consecutive f32 per thread.
  const int arow = tid >> 2;
  const int akk  = (tid & 3) * 8;
  // B staging slot: 128 n x 32 k; per thread n fixed, k = kb0 + it contiguous.
  const int bn  = tid & 127;
  const int kb0 = (tid >> 7) * 16;
  const int gcol = col0 + bn;
  const int h    = gcol / 96;
  const int w    = gcol - h * 96;
  const size_t xbase = (size_t)bz * Cin * 9216;
  // Loop-invariant im2col helpers: 9-bit validity mask + base offset.
  const int pbase = (h - 1) * 96 + (w - 1);
  unsigned mask = 0;
#pragma unroll
  for (int rr = 0; rr < 3; ++rr)
#pragma unroll
    for (int ss = 0; ss < 3; ++ss)
      if ((unsigned)(h + rr - 1) < 96u && (unsigned)(w + ss - 1) < 96u)
        mask |= 1u << (rr * 3 + ss);

  float ra[8];
  float rb[16];

  auto gatherA = [&](int k0) {
    const float* src = Wt + (size_t)(m0 + arow) * Ktot + k0 + akk;
#pragma unroll
    for (int j = 0; j < 8; ++j) ra[j] = src[j];
  };
  // Branchless halo handling: clamp offset via cndmask, load unconditionally,
  // then select 0.0 -- no exec-mask branches in the hot gather.
  auto gatherB = [&](int k0) {
#pragma unroll
    for (int it = 0; it < 16; ++it) {
      int kg = k0 + kb0 + it;
      int ci = kg / 9;
      int rs = kg - ci * 9;
      int r  = rs / 3;
      int s  = rs - r * 3;
      bool valid = (mask >> rs) & 1u;
      int  off   = valid ? (pbase + r * 96 + s) : 0;
      float ld   = X[xbase + (size_t)ci * 9216 + off];
      rb[it] = valid ? ld : 0.0f;
    }
  };

  v8f acc[2][2] = {};

  gatherA(0);
  gatherB(0);
  int p = 0;
  for (int k0 = 0; k0 < Ktot; k0 += 32) {
    store_pk8(&As[p][arow * LSTR + akk], ra);
    store_pk8(&Bs[p][bn * LSTR + kb0], rb);
    store_pk8(&Bs[p][bn * LSTR + kb0 + 8], rb + 8);
    __syncthreads();
    int kn = k0 + 32;
    if (kn < Ktot) { gatherA(kn); gatherB(kn); }  // overlap with WMMAs below
    v16h a0 = frag_a(&As[p][(wm * 32 + lrow) * LSTR + asel]);
    v16h a1 = frag_a(&As[p][(wm * 32 + 16 + lrow) * LSTR + asel]);
    v16h f0 = frag_b(&Bs[p][(wn * 32 + lrow) * LSTR + bsel]);
    v16h f1 = frag_b(&Bs[p][(wn * 32 + 16 + lrow) * LSTR + bsel]);
    acc[0][0] = wmma_f16(a0, f0, acc[0][0]);
    acc[0][1] = wmma_f16(a0, f1, acc[0][1]);
    acc[1][0] = wmma_f16(a1, f0, acc[1][0]);
    acc[1][1] = wmma_f16(a1, f1, acc[1][1]);
    p ^= 1;  // next chunk stages into the other buffer: no trailing barrier
  }

  const int msel = (lane & 16) ? 8 : 0;
#pragma unroll
  for (int sm = 0; sm < 2; ++sm) {
#pragma unroll
    for (int r = 0; r < 8; ++r) {
      int m = m0 + wm * 32 + sm * 16 + msel + r;
      float sc = scale[m], bi = bias[m];
      size_t base = ((size_t)bz * Cout + m) * 9216;
      int n0 = col0 + wn * 32 + lrow;
      Y[base + n0]      = fmaxf(acc[sm][0][r] * sc + bi, 0.0f);
      Y[base + n0 + 16] = fmaxf(acc[sm][1][r] * sc + bi, 0.0f);
    }
  }
}

// ---------------------------------------------------------------------------
// K3: build t = concat(bilinear(maxpool2(x1)), x2, bilinear(avgpool2(x3)))
// ---------------------------------------------------------------------------
__device__ __forceinline__ float poolcell(const float* Yc, int ph, int pw, bool ismax) {
  float a = Yc[(2 * ph) * 96 + 2 * pw];
  float b = Yc[(2 * ph) * 96 + 2 * pw + 1];
  float c = Yc[(2 * ph + 1) * 96 + 2 * pw];
  float d = Yc[(2 * ph + 1) * 96 + 2 * pw + 1];
  return ismax ? fmaxf(fmaxf(a, b), fmaxf(c, d)) : 0.25f * (a + b + c + d);
}

__global__ void branch_build(const float* __restrict__ Y, float* __restrict__ T) {
  int idx = blockIdx.x * 256 + threadIdx.x;
  if (idx >= 4 * 192 * 9216) return;
  int w = idx % 96;
  int h = (idx / 96) % 96;
  int c = (idx / 9216) % 192;
  int b = idx / (9216 * 192);
  const float* Yc = Y + ((size_t)(b * 192 + c)) * 9216;
  float val;
  if (c >= 64 && c < 128) {
    val = Yc[h * 96 + w];
  } else {
    bool ismax = (c < 64);
    float ch = fminf(fmaxf(0.5f * h - 0.25f, 0.0f), 47.0f);
    float cw = fminf(fmaxf(0.5f * w - 0.25f, 0.0f), 47.0f);
    int hlo = (int)ch; int hhi = min(hlo + 1, 47); float hf = ch - hlo;
    int wlo = (int)cw; int whi = min(wlo + 1, 47); float wf = cw - wlo;
    float p00 = poolcell(Yc, hlo, wlo, ismax);
    float p01 = poolcell(Yc, hlo, whi, ismax);
    float p10 = poolcell(Yc, hhi, wlo, ismax);
    float p11 = poolcell(Yc, hhi, whi, ismax);
    val = (1.0f - hf) * ((1.0f - wf) * p00 + wf * p01) +
          hf * ((1.0f - wf) * p10 + wf * p11);
  }
  T[idx] = val;
}

// ---------------------------------------------------------------------------
// K5: inverse L2 norms of q,k rows (length 9216).
// ---------------------------------------------------------------------------
__global__ __launch_bounds__(256) void l2norm_inv(const float* __restrict__ T2,
                                                  float* __restrict__ qin,
                                                  float* __restrict__ kin) {
  int c = blockIdx.x, b = blockIdx.y, isk = blockIdx.z;
  const float* row = T2 + ((size_t)(b * 576 + isk * 192 + c)) * 9216;
  __shared__ float red[256];
  float s = 0.0f;
  for (int i = threadIdx.x; i < 9216; i += 256) { float v = row[i]; s += v * v; }
  red[threadIdx.x] = s;
  __syncthreads();
  for (int st = 128; st > 0; st >>= 1) {
    if (threadIdx.x < st) red[threadIdx.x] += red[threadIdx.x + st];
    __syncthreads();
  }
  if (threadIdx.x == 0) {
    float inv = 1.0f / fmaxf(sqrtf(red[0]), 1e-12f);
    (isk ? kin : qin)[b * 192 + c] = inv;
  }
}

// ---------------------------------------------------------------------------
// K6: attn = (q_norm @ k_norm^T) * temp.  M=N=192, K=9216.  Both operands are
// row-major [c][n] => both stage directly as [row][k] tiles.  64x64 tile,
// pipelined, double-buffered (288 K-chunks, one barrier each).
// ---------------------------------------------------------------------------
__global__ __launch_bounds__(256) void attn_qk_wmma(
    const float* __restrict__ T2, const float* __restrict__ qin,
    const float* __restrict__ kin, const float* __restrict__ temp,
    float* __restrict__ attn) {
  __shared__ _Float16 As[2][64 * LSTR];
  __shared__ _Float16 Bs[2][64 * LSTR];
  const int b = blockIdx.z, i0 = blockIdx.y * 64, j0 = blockIdx.x * 64;
  const int tid = threadIdx.x;
  const int lane = tid & 31;
  const int wave = tid >> 5;
  const int wm = wave & 3, wn = wave >> 2;
  const int lrow = lane & 15;
  const int asel = (lane & 16) ? 16 : 0;
  const int bsel = (lane & 16) ? 16 : 0;
  const int row = tid >> 2;
  const int kk  = (tid & 3) * 8;
  const float* qsrc = T2 + ((size_t)(b * 576 + i0 + row)) * 9216 + kk;
  const float* ksrc = T2 + ((size_t)(b * 576 + 192 + j0 + row)) * 9216 + kk;

  float ra[8], rb[8];
  auto gather = [&](int k0) {
#pragma unroll
    for (int j = 0; j < 8; ++j) {
      ra[j] = qsrc[k0 + j];
      rb[j] = ksrc[k0 + j];
    }
  };

  v8f acc0 = {}; v8f acc1 = {};
  gather(0);
  int p = 0;
  for (int k0 = 0; k0 < 9216; k0 += 32) {
    store_pk8(&As[p][row * LSTR + kk], ra);
    store_pk8(&Bs[p][row * LSTR + kk], rb);
    __syncthreads();
    if (k0 + 32 < 9216) gather(k0 + 32);  // overlap with WMMAs
    v16h a  = frag_a(&As[p][(wm * 16 + lrow) * LSTR + asel]);
    v16h f0 = frag_b(&Bs[p][(wn * 32 + lrow) * LSTR + bsel]);
    v16h f1 = frag_b(&Bs[p][(wn * 32 + 16 + lrow) * LSTR + bsel]);
    acc0 = wmma_f16(a, f0, acc0);
    acc1 = wmma_f16(a, f1, acc1);
    p ^= 1;
  }
  const int msel = (lane & 16) ? 8 : 0;
  float tsc = temp[0];
#pragma unroll
  for (int r = 0; r < 8; ++r) {
    int i = i0 + wm * 16 + msel + r;
    int j = j0 + wn * 32 + lrow;
    float qi = qin[b * 192 + i] * tsc;
    size_t base = ((size_t)b * 192 + i) * 192;
    attn[base + j]      = acc0[r] * qi * kin[b * 192 + j];
    attn[base + j + 16] = acc1[r] * qi * kin[b * 192 + j + 16];
  }
}

// ---------------------------------------------------------------------------
// K7: row softmax, rows of 192.
// ---------------------------------------------------------------------------
__global__ __launch_bounds__(256) void softmax192(float* __restrict__ attn) {
  int b = blockIdx.y, i = blockIdx.x;
  float* row = attn + ((size_t)(b * 192 + i)) * 192;
  __shared__ float red[256];
  int t = threadIdx.x;
  float v = (t < 192) ? row[t] : -3.0e38f;
  red[t] = v;
  __syncthreads();
  for (int s = 128; s > 0; s >>= 1) {
    if (t < s) red[t] = fmaxf(red[t], red[t + s]);
    __syncthreads();
  }
  float mx = red[0];
  __syncthreads();
  float e = (t < 192) ? __expf(v - mx) : 0.0f;
  red[t] = e;
  __syncthreads();
  for (int s = 128; s > 0; s >>= 1) {
    if (t < s) red[t] += red[t + s];
    __syncthreads();
  }
  float inv = 1.0f / red[0];
  if (t < 192) row[t] = e * inv;
}

// ---------------------------------------------------------------------------
// K8: O = attn @ v.  M=192, N=9216, K=192.  64x128 tile, pipelined,
// double-buffered. A row-major [M][K]; B (=v) is [K][N] so it stages
// transposed into Bs[n][k] (contiguous 16-k slice per thread -> b128 stores).
// ---------------------------------------------------------------------------
__global__ __launch_bounds__(256) void attn_v_wmma(
    const float* __restrict__ attn, const float* __restrict__ T2,
    float* __restrict__ O) {
  __shared__ _Float16 As[2][64 * LSTR];
  __shared__ _Float16 Bs[2][128 * LSTR];
  const int b = blockIdx.z, m0 = blockIdx.y * 64, col0 = blockIdx.x * 128;
  const int tid = threadIdx.x;
  const int lane = tid & 31;
  const int wave = tid >> 5;
  const int wm = wave & 1, wn = wave >> 1;
  const int lrow = lane & 15;
  const int asel = (lane & 16) ? 16 : 0;
  const int bsel = (lane & 16) ? 16 : 0;
  const int arow = tid >> 2;
  const int akk  = (tid & 3) * 8;
  const int bn  = tid & 127;
  const int kb0 = (tid >> 7) * 16;

  float ra[8], rb[16];
  auto gatherA = [&](int k0) {
    const float* src = attn + ((size_t)b * 192 + m0 + arow) * 192 + k0 + akk;
#pragma unroll
    for (int j = 0; j < 8; ++j) ra[j] = src[j];
  };
  auto gatherB = [&](int k0) {
#pragma unroll
    for (int it = 0; it < 16; ++it) {
      int d = k0 + kb0 + it;
      rb[it] = T2[((size_t)(b * 576 + 384 + d)) * 9216 + col0 + bn];
    }
  };

  v8f acc[2][2] = {};
  gatherA(0);
  gatherB(0);
  int p = 0;
  for (int k0 = 0; k0 < 192; k0 += 32) {
    store_pk8(&As[p][arow * LSTR + akk], ra);
    store_pk8(&Bs[p][bn * LSTR + kb0], rb);
    store_pk8(&Bs[p][bn * LSTR + kb0 + 8], rb + 8);
    __syncthreads();
    if (k0 + 32 < 192) { gatherA(k0 + 32); gatherB(k0 + 32); }
    v16h a0 = frag_a(&As[p][(wm * 32 + lrow) * LSTR + asel]);
    v16h a1 = frag_a(&As[p][(wm * 32 + 16 + lrow) * LSTR + asel]);
    v16h f0 = frag_b(&Bs[p][(wn * 32 + lrow) * LSTR + bsel]);
    v16h f1 = frag_b(&Bs[p][(wn * 32 + 16 + lrow) * LSTR + bsel]);
    acc[0][0] = wmma_f16(a0, f0, acc[0][0]);
    acc[0][1] = wmma_f16(a0, f1, acc[0][1]);
    acc[1][0] = wmma_f16(a1, f0, acc[1][0]);
    acc[1][1] = wmma_f16(a1, f1, acc[1][1]);
    p ^= 1;
  }
  const int msel = (lane & 16) ? 8 : 0;
#pragma unroll
  for (int sm = 0; sm < 2; ++sm) {
#pragma unroll
    for (int r = 0; r < 8; ++r) {
      int m = m0 + wm * 32 + sm * 16 + msel + r;
      size_t base = ((size_t)b * 192 + m) * 9216;
      int n0 = col0 + wn * 32 + lrow;
      O[base + n0]      = acc[sm][0][r];
      O[base + n0 + 16] = acc[sm][1][r];
    }
  }
}

// ---------------------------------------------------------------------------
// K9: depthwise 3x3 conv + bias (groups = 192), branchless halo.
// ---------------------------------------------------------------------------
__global__ void dwconv3x3(const float* __restrict__ O, const float* __restrict__ W2,
                          const float* __restrict__ B2, float* __restrict__ out) {
  int idx = blockIdx.x * 256 + threadIdx.x;
  if (idx >= 4 * 192 * 9216) return;
  int w = idx % 96;
  int h = (idx / 96) % 96;
  int c = (idx / 9216) % 192;
  int b = idx / (9216 * 192);
  const float* Oc = O + ((size_t)(b * 192 + c)) * 9216;
  const float* wp = W2 + c * 9;
  float acc = B2[c];
#pragma unroll
  for (int r = 0; r < 3; ++r) {
#pragma unroll
    for (int s = 0; s < 3; ++s) {
      int hh = h + r - 1, ww = w + s - 1;
      bool valid = ((unsigned)hh < 96u) && ((unsigned)ww < 96u);
      int off = valid ? (hh * 96 + ww) : 0;
      float ld = Oc[off];
      acc += (valid ? ld : 0.0f) * wp[r * 3 + s];
    }
  }
  out[idx] = acc;
}

// ---------------------------------------------------------------------------
// Host-side orchestration.
// Workspace layout (bytes):
//   [0)            Y   : conv0 out [4,192,96,96] f32  (28,311,552) — reused as O
//   [28,311,552)   T   : branch-mixed tensor           (28,311,552)
//   [56,623,104)   T2  : conv1 out [4,576,96,96] f32   (84,934,656)
//   [141,557,760)  ATT : attention [4,192,192] f32     (   589,824)
//   [142,147,584)  PAR : folded BN params (1536 f32)
//   [142,153,728)  NRM : qin/kin inv-norms (1536 f32)
// ---------------------------------------------------------------------------
extern "C" void kernel_launch(void* const* d_in, const int* in_sizes, int n_in,
                              void* d_out, int out_size, void* d_ws, size_t ws_size,
                              hipStream_t stream) {
  const float* x   = (const float*)d_in[0];
  const float* w0  = (const float*)d_in[1];
  const float* b0  = (const float*)d_in[2];
  const float* g0  = (const float*)d_in[3];
  const float* be0 = (const float*)d_in[4];
  const float* m0  = (const float*)d_in[5];
  const float* v0  = (const float*)d_in[6];
  const float* w1  = (const float*)d_in[7];
  const float* b1  = (const float*)d_in[8];
  const float* g1  = (const float*)d_in[9];
  const float* be1 = (const float*)d_in[10];
  const float* m1  = (const float*)d_in[11];
  const float* v1  = (const float*)d_in[12];
  const float* tmp = (const float*)d_in[13];
  const float* w2  = (const float*)d_in[14];
  const float* b2  = (const float*)d_in[15];

  char* ws = (char*)d_ws;
  float* Y   = (float*)(ws);
  float* T   = (float*)(ws + 28311552u);
  float* T2  = (float*)(ws + 56623104u);
  float* ATT = (float*)(ws + 141557760u);
  float* PAR = (float*)(ws + 142147584u);
  float* NRM = (float*)(ws + 142153728u);

  const int TOTAL = 4 * 192 * 9216;

  fold_bn<<<3, 256, 0, stream>>>(b0, g0, be0, m0, v0, b1, g1, be1, m1, v1, PAR);
  conv3x3_wmma<<<dim3(72, 3, 4), 256, 0, stream>>>(x, w0, PAR, PAR + 192, Y, 64, 192);
  branch_build<<<(TOTAL + 255) / 256, 256, 0, stream>>>(Y, T);
  conv3x3_wmma<<<dim3(72, 9, 4), 256, 0, stream>>>(T, w1, PAR + 384, PAR + 960, T2, 192, 576);
  l2norm_inv<<<dim3(192, 4, 2), 256, 0, stream>>>(T2, NRM, NRM + 768);
  attn_qk_wmma<<<dim3(3, 3, 4), 256, 0, stream>>>(T2, NRM, NRM + 768, tmp, ATT);
  softmax192<<<dim3(192, 4), 256, 0, stream>>>(ATT);
  attn_v_wmma<<<dim3(72, 3, 4), 256, 0, stream>>>(ATT, T2, Y);  // Y reused as O
  dwconv3x3<<<(TOTAL + 255) / 256, 256, 0, stream>>>(Y, w2, b2, (float*)d_out);
}
